// MambaOnlyLayer_42812234006552
// MI455X (gfx1250) — compile-verified, matrix operand order checked by reading, not run on hardware
//
#include <hip/hip_runtime.h>
#include <hip/hip_bf16.h>
#include <math.h>

// ---------------------------------------------------------------------------
// Mamba2 decoder layer for MI455X (gfx1250, wave32, WMMA).
// All large GEMMs (in_proj / out_proj / FFN / SSD chunk matmuls) run through
// v_wmma_f32_16x16x32_bf16 with f32 accumulation. GEMM LDS staging uses
// gfx1250 async global->LDS copies (ASYNCcnt) with double buffering when the
// toolchain exposes the builtins.
// ---------------------------------------------------------------------------

typedef __bf16 v16bf __attribute__((ext_vector_type(16)));
typedef __bf16 v8bf  __attribute__((ext_vector_type(8)));
typedef float  v8f   __attribute__((ext_vector_type(8)));
typedef int    v4i   __attribute__((vector_size(16)));   // matches builtin param type

#define DEV static __device__ __forceinline__

#ifndef __has_builtin
#define __has_builtin(x) 0
#endif
#if __has_builtin(__builtin_amdgcn_global_load_async_to_lds_b128) && \
    __has_builtin(__builtin_amdgcn_s_wait_asynccnt)
#define HAVE_ASYNC_LDS 1
#else
#define HAVE_ASYNC_LDS 0
#endif

constexpr int Bb     = 8;
constexpr int S      = 2048;
constexpr int T      = Bb * S;          // 16384 tokens
constexpr int D      = 512;
constexpr int DIN    = 1024;
constexpr int DSTATE = 128;
constexpr int CONVD  = 1280;            // DIN + 2*DSTATE
constexpr int DIP    = 2320;            // 2*DIN + 2*DSTATE + NH
constexpr int NH     = 16;
constexpr int HD     = 64;
constexpr int DCONV  = 4;
constexpr int FFD    = 2048;
constexpr int CHK    = 128;
constexpr int NCH    = S / CHK;         // 16
constexpr float EPSF = 1e-5f;

// ----- WMMA helpers ---------------------------------------------------------

DEV v16bf ld_frag(const __bf16* tile, int row0, int lane) {
  // tile is (rows x 32) row-major bf16 in LDS.
  // lane group 0-15: rows 0..15, K halves {0..7, 16..23}
  // lane group 16-31: same rows, K halves {8..15, 24..31}
  const __bf16* p = tile + (row0 + (lane & 15)) * 32 + ((lane >> 4) * 8);
  union { v16bf v; v8bf h[2]; } u;
  u.h[0] = *(const v8bf*)(p);
  u.h[1] = *(const v8bf*)(p + 16);
  return u.v;
}

DEV v8f wmma_bf16(v16bf a, v16bf b, v8f c) {
  return __builtin_amdgcn_wmma_f32_16x16x32_bf16(false, a, false, b,
                                                 (short)0, c, false, false);
}

#if HAVE_ASYNC_LDS
DEV void async_cp16(__bf16* lds_dst, const __bf16* gsrc) {
  __builtin_amdgcn_global_load_async_to_lds_b128(
      (__attribute__((address_space(1))) v4i*)(gsrc),
      (__attribute__((address_space(3))) v4i*)(lds_dst),
      0, 0);
}
#endif

DEV float block_sum(float v, float* red) {
  #pragma unroll
  for (int o = 16; o > 0; o >>= 1) v += __shfl_down(v, o, 32);
  __syncthreads();
  if ((threadIdx.x & 31) == 0) red[threadIdx.x >> 5] = v;
  __syncthreads();
  float t = 0.f;
  #pragma unroll
  for (int i = 0; i < 8; i++) t += red[i];
  return t;
}

// ----- elementwise ----------------------------------------------------------

__global__ void cvt_f32_bf16_kernel(const float* __restrict__ in,
                                    __bf16* __restrict__ out, size_t n) {
  size_t i = (size_t)blockIdx.x * blockDim.x + threadIdx.x;
  if (i < n) out[i] = (__bf16)in[i];
}

// causal depthwise conv1d + SiLU over zxbcdt[:, :, 1024:2304]; routes to x/B/C (bf16)
__global__ void conv_silu_kernel(const float* __restrict__ zxb,
                                 const float* __restrict__ cw,
                                 const float* __restrict__ cb,
                                 __bf16* __restrict__ xbf,
                                 __bf16* __restrict__ Bbf,
                                 __bf16* __restrict__ Cbf) {
  size_t idx = (size_t)blockIdx.x * 256 + threadIdx.x;
  if (idx >= (size_t)T * CONVD) return;
  int ch = (int)(idx % CONVD);
  size_t tok = idx / CONVD;
  int s = (int)(tok % S);
  float acc = cb[ch];
  #pragma unroll
  for (int k = 0; k < DCONV; k++) {
    int sp = s - (DCONV - 1) + k;
    if (sp >= 0)
      acc += zxb[(tok + (size_t)(sp - s)) * DIP + DIN + ch] * cw[ch * DCONV + k];
  }
  float v = acc / (1.f + expf(-acc));         // SiLU
  if (ch < DIN)                 xbf[tok * DIN + ch] = (__bf16)v;
  else if (ch < DIN + DSTATE)   Bbf[tok * DSTATE + (ch - DIN)] = (__bf16)v;
  else                          Cbf[tok * DSTATE + (ch - DIN - DSTATE)] = (__bf16)v;
}

__global__ void dt_softplus_kernel(const float* __restrict__ zxb,
                                   const float* __restrict__ dtbias,
                                   float* __restrict__ dtb) {
  int idx = blockIdx.x * 256 + threadIdx.x;
  if (idx >= T * NH) return;
  int h = idx & (NH - 1);
  size_t tok = (size_t)idx >> 4;
  float x = zxb[tok * DIP + DIN + CONVD + h] + dtbias[h];
  dtb[idx] = (x > 20.f) ? x : log1pf(expf(x));
}

// cumulative sum of dt*A within each chunk; one thread per (b,c,h)
__global__ void dta_cumsum_kernel(const float* __restrict__ dtb,
                                  const float* __restrict__ Alog,
                                  float* __restrict__ acs) {
  int idx = blockIdx.x * 256 + threadIdx.x;
  if (idx >= Bb * NCH * NH) return;
  int h = idx & (NH - 1);
  int bc = idx >> 4;
  float Ah = -expf(Alog[h]);
  float acc = 0.f;
  for (int i = 0; i < CHK; i++) {
    size_t sidx = (size_t)(bc * CHK + i) * NH + h;
    acc += dtb[sidx] * Ah;
    acs[sidx] = acc;
  }
}

// ----- generic WMMA GEMM: C[m,n] = sum_k A[m,k]*W[n,k] (+bias) --------------
// MODE 0: f32 output; MODE 1: exact GELU -> bf16 output.
// Out-of-range N rows of W are clamped to row 0: their garbage only reaches
// output columns >= N, which are never stored.

template <int MODE>
__global__ __launch_bounds__(256)
void gemm_nt_kernel(const __bf16* __restrict__ A, const __bf16* __restrict__ W,
                    const float* __restrict__ bias, void* __restrict__ out,
                    int M, int N, int K) {
#if HAVE_ASYNC_LDS
  __shared__ __align__(16) __bf16 As[2][128 * 32];
  __shared__ __align__(16) __bf16 Bs[2][128 * 32];
#else
  __shared__ __align__(16) __bf16 As[1][128 * 32];
  __shared__ __align__(16) __bf16 Bs[1][128 * 32];
#endif
  const int tid  = threadIdx.x;
  const int lane = tid & 31;
  const int wave = tid >> 5;                  // 8 waves
  const int wm   = (wave & 1) * 64;
  const int wn   = (wave >> 1) * 32;
  const int m0   = blockIdx.y * 128;
  const int n0   = blockIdx.x * 128;
  const int nk   = K >> 5;

  v8f acc[4][2];
  #pragma unroll
  for (int a = 0; a < 4; a++)
    #pragma unroll
    for (int b = 0; b < 2; b++)
      #pragma unroll
      for (int q = 0; q < 8; q++) acc[a][b][q] = 0.f;

#if HAVE_ASYNC_LDS
  auto issue = [&](int buf, int k0) {
    #pragma unroll
    for (int i = 0; i < 2; i++) {
      int s  = tid + i * 256;                 // 512 slots of 8 bf16
      int r  = s >> 2;
      int c8 = (s & 3) * 8;
      async_cp16(&As[buf][r * 32 + c8], A + (size_t)(m0 + r) * K + k0 + c8);
      int n = n0 + r;
      if (n >= N) n = 0;                      // clamp (see note above)
      async_cp16(&Bs[buf][r * 32 + c8], W + (size_t)n * K + k0 + c8);
    }
  };
  issue(0, 0);
#endif

  for (int kb = 0; kb < nk; kb++) {
#if HAVE_ASYNC_LDS
    const int p = kb & 1;
    __builtin_amdgcn_s_wait_asynccnt(0);      // my copies into buffer p done
    __syncthreads();                          // everyone's copies done; buffer 1-p free
    if (kb + 1 < nk) issue(1 - p, (kb + 1) * 32);
#else
    const int p = 0;
    const int k0 = kb * 32;
    __syncthreads();
    #pragma unroll
    for (int i = 0; i < 2; i++) {
      int s  = tid + i * 256;
      int r  = s >> 2;
      int c8 = (s & 3) * 8;
      *(v8bf*)&As[0][r * 32 + c8] = *(const v8bf*)(A + (size_t)(m0 + r) * K + k0 + c8);
      int n = n0 + r;
      if (n >= N) n = 0;                      // clamp (see note above)
      *(v8bf*)&Bs[0][r * 32 + c8] = *(const v8bf*)(W + (size_t)n * K + k0 + c8);
    }
    __syncthreads();
#endif
    v16bf af[4], wf[2];
    #pragma unroll
    for (int mi = 0; mi < 4; mi++) af[mi] = ld_frag(As[p], wm + mi * 16, lane);
    #pragma unroll
    for (int ni = 0; ni < 2; ni++) wf[ni] = ld_frag(Bs[p], wn + ni * 16, lane);
    #pragma unroll
    for (int mi = 0; mi < 4; mi++)
      #pragma unroll
      for (int ni = 0; ni < 2; ni++)
        acc[mi][ni] = wmma_bf16(af[mi], wf[ni], acc[mi][ni]);
  }

  const int rsub = (lane >> 4) * 8;
  const int csub = lane & 15;
  #pragma unroll
  for (int mi = 0; mi < 4; mi++)
    #pragma unroll
    for (int ni = 0; ni < 2; ni++)
      #pragma unroll
      for (int r = 0; r < 8; r++) {
        int gm = m0 + wm + mi * 16 + rsub + r;
        int gn = n0 + wn + ni * 16 + csub;
        if (gn < N) {
          float v = acc[mi][ni][r];
          if (bias) v += bias[gn];
          if (MODE == 0) {
            ((float*)out)[(size_t)gm * N + gn] = v;
          } else {
            float g = 0.5f * v * (1.0f + erff(v * 0.70710678118654752f));
            ((__bf16*)out)[(size_t)gm * N + gn] = (__bf16)g;
          }
        }
      }
}

// ----- SSD: per-chunk scores G = C @ B^T (128x128x128) ----------------------

__global__ __launch_bounds__(256)
void chunk_scores_kernel(const __bf16* __restrict__ Cbf,
                         const __bf16* __restrict__ Bbf,
                         float* __restrict__ G) {
  const int bc = blockIdx.x;                  // b*NCH + c
  const __bf16* Abase = Cbf + (size_t)bc * CHK * DSTATE;
  const __bf16* Bbase = Bbf + (size_t)bc * CHK * DSTATE;
  float* Gout = G + (size_t)bc * CHK * CHK;

  __shared__ __align__(16) __bf16 As[128 * 32];
  __shared__ __align__(16) __bf16 Bs[128 * 32];
  const int tid = threadIdx.x, lane = tid & 31, wave = tid >> 5;
  const int wm = (wave & 1) * 64, wn = (wave >> 1) * 32;

  v8f acc[4][2];
  #pragma unroll
  for (int a = 0; a < 4; a++)
    #pragma unroll
    for (int b = 0; b < 2; b++)
      #pragma unroll
      for (int q = 0; q < 8; q++) acc[a][b][q] = 0.f;

  for (int k0 = 0; k0 < DSTATE; k0 += 32) {
    __syncthreads();
#if HAVE_ASYNC_LDS
    #pragma unroll
    for (int i = 0; i < 2; i++) {
      int s = tid + i * 256;
      int r = s >> 2, c8 = (s & 3) * 8;
      async_cp16(&As[r * 32 + c8], Abase + (size_t)r * DSTATE + k0 + c8);
      async_cp16(&Bs[r * 32 + c8], Bbase + (size_t)r * DSTATE + k0 + c8);
    }
    __builtin_amdgcn_s_wait_asynccnt(0);
#else
    #pragma unroll
    for (int i = 0; i < 2; i++) {
      int s = tid + i * 256;
      int r = s >> 2, c8 = (s & 3) * 8;
      *(v8bf*)&As[r * 32 + c8] = *(const v8bf*)(Abase + (size_t)r * DSTATE + k0 + c8);
      *(v8bf*)&Bs[r * 32 + c8] = *(const v8bf*)(Bbase + (size_t)r * DSTATE + k0 + c8);
    }
#endif
    __syncthreads();
    v16bf af[4], wf[2];
    #pragma unroll
    for (int mi = 0; mi < 4; mi++) af[mi] = ld_frag(As, wm + mi * 16, lane);
    #pragma unroll
    for (int ni = 0; ni < 2; ni++) wf[ni] = ld_frag(Bs, wn + ni * 16, lane);
    #pragma unroll
    for (int mi = 0; mi < 4; mi++)
      #pragma unroll
      for (int ni = 0; ni < 2; ni++)
        acc[mi][ni] = wmma_bf16(af[mi], wf[ni], acc[mi][ni]);
  }
  const int rsub = (lane >> 4) * 8, csub = lane & 15;
  #pragma unroll
  for (int mi = 0; mi < 4; mi++)
    #pragma unroll
    for (int ni = 0; ni < 2; ni++)
      #pragma unroll
      for (int r = 0; r < 8; r++) {
        int gm = wm + mi * 16 + rsub + r;
        int gn = wn + ni * 16 + csub;
        Gout[(size_t)gm * CHK + gn] = acc[mi][ni][r];
      }
}

// ----- SSD: per-(b,c,h) Y_diag = (G*L*dt) @ X  and  states = (B^T w) @ X ----

__global__ __launch_bounds__(256)
void ydiag_states_kernel(const float* __restrict__ G, const float* __restrict__ acs,
                         const float* __restrict__ dtb, const __bf16* __restrict__ xbf,
                         const __bf16* __restrict__ Bbf, const float* __restrict__ Dhead,
                         float* __restrict__ y, float* __restrict__ st) {
  const int id = blockIdx.x;                  // (b*NCH + c)*NH + h
  const int h  = id & (NH - 1);
  const int bc = id >> 4;
  const int tid = threadIdx.x, lane = tid & 31, wave = tid >> 5;

  __shared__ float acsS[CHK], dtS[CHK], wS[CHK];
  __shared__ __align__(16) __bf16 Ms[CHK * 32];     // masked decay matrix slice
  __shared__ __align__(16) __bf16 Bt[DSTATE * 32];  // weighted B^T slice
  __shared__ __align__(16) __bf16 Xt[HD * 32];      // X^T slice (shared B-operand)

  if (tid < CHK) {
    size_t sidx = (size_t)(bc * CHK + tid) * NH + h;
    acsS[tid] = acs[sidx];
    dtS[tid]  = dtb[sidx];
  }
  __syncthreads();
  if (tid < CHK) wS[tid] = expf(acsS[CHK - 1] - acsS[tid]) * dtS[tid];

  v8f accY[4], accS[4];
  #pragma unroll
  for (int a = 0; a < 4; a++)
    #pragma unroll
    for (int q = 0; q < 8; q++) { accY[a][q] = 0.f; accS[a][q] = 0.f; }

  const float*  Gb  = G   + (size_t)bc * CHK * CHK;
  const __bf16* Bw  = Bbf + (size_t)bc * CHK * DSTATE;
  const __bf16* Xb  = xbf + (size_t)bc * CHK * DIN + h * HD;

  for (int k0 = 0; k0 < CHK; k0 += 32) {
    __syncthreads();
    #pragma unroll
    for (int i = 0; i < 16; i++) {            // Ms: 128x32
      int e = tid + i * 256;
      int r = e >> 5, jj = e & 31, j = k0 + jj;
      float g = Gb[(size_t)r * CHK + j];
      float v = (j <= r) ? g * expf(acsS[r] - acsS[j]) * dtS[j] : 0.f;
      Ms[r * 32 + jj] = (__bf16)v;
    }
    #pragma unroll
    for (int i = 0; i < 16; i++) {            // Bt: 128x32 (n-major)
      int e = tid + i * 256;
      int n = e & 127, jj = e >> 7, j = k0 + jj;
      Bt[n * 32 + jj] = (__bf16)((float)Bw[(size_t)j * DSTATE + n] * wS[j]);
    }
    #pragma unroll
    for (int i = 0; i < 8; i++) {             // Xt: 64x32 (p-major)
      int e = tid + i * 256;
      int p = e & 63, jj = e >> 6, j = k0 + jj;
      Xt[p * 32 + jj] = Xb[(size_t)j * DIN + p];
    }
    __syncthreads();
    v16bf aY = ld_frag(Ms, wave * 16, lane);
    v16bf aS = ld_frag(Bt, wave * 16, lane);
    #pragma unroll
    for (int ni = 0; ni < 4; ni++) {
      v16bf bx = ld_frag(Xt, ni * 16, lane);
      accY[ni] = wmma_bf16(aY, bx, accY[ni]);
      accS[ni] = wmma_bf16(aS, bx, accS[ni]);
    }
  }

  const float Dh = Dhead[h];
  const int rsub = (lane >> 4) * 8, csub = lane & 15;
  float* yb  = y  + (size_t)bc * CHK * DIN + h * HD;
  float* stb = st + (size_t)id * HD * DSTATE;       // (b,c,h,p,n)
  #pragma unroll
  for (int ni = 0; ni < 4; ni++)
    #pragma unroll
    for (int r = 0; r < 8; r++) {
      int row = wave * 16 + rsub + r;         // i for Y, n for states
      int p   = ni * 16 + csub;
      float xv = (float)Xb[(size_t)row * DIN + p];
      yb[(size_t)row * DIN + p] = accY[ni][r] + Dh * xv;
      stb[(size_t)p * DSTATE + row] = accS[ni][r];
    }
}

// ----- SSD: sequential inter-chunk scan (in-place states -> prev) -----------

__global__ __launch_bounds__(256)
void scan_kernel(const float* __restrict__ acs, float* __restrict__ st) {
  const int b = blockIdx.x >> 4, h = blockIdx.x & (NH - 1);
  float h0[32];
  #pragma unroll
  for (int i = 0; i < 32; i++) h0[i] = 0.f;
  for (int c = 0; c < NCH; c++) {
    const float cd = expf(acs[((size_t)(b * NCH + c) * CHK + CHK - 1) * NH + h]);
    float* base = st + ((size_t)(b * NCH + c) * NH + h) * HD * DSTATE;
    #pragma unroll
    for (int i = 0; i < 32; i++) {
      size_t e = (size_t)threadIdx.x + (size_t)i * 256;
      float sv = base[e];
      base[e] = h0[i];                        // prev = carry at chunk start
      h0[i] = cd * h0[i] + sv;
    }
  }
}

// ----- SSD: Y_off = (C * exp(acs)) @ prev^T, accumulated into y -------------

__global__ __launch_bounds__(256)
void yoff_kernel(const float* __restrict__ acs, const __bf16* __restrict__ Cbf,
                 const float* __restrict__ prev, float* __restrict__ y) {
  const int id = blockIdx.x;
  const int h = id & (NH - 1), bc = id >> 4;
  const int tid = threadIdx.x, lane = tid & 31, wave = tid >> 5;

  __shared__ float eS[CHK];
  __shared__ __align__(16) __bf16 Cs[CHK * 32];
  __shared__ __align__(16) __bf16 Pv[HD * 32];

  if (tid < CHK) eS[tid] = expf(acs[(size_t)(bc * CHK + tid) * NH + h]);

  v8f acc[4];
  #pragma unroll
  for (int a = 0; a < 4; a++)
    #pragma unroll
    for (int q = 0; q < 8; q++) acc[a][q] = 0.f;

  const __bf16* Cb = Cbf + (size_t)bc * CHK * DSTATE;
  const float*  Pb = prev + (size_t)id * HD * DSTATE;

  for (int k0 = 0; k0 < DSTATE; k0 += 32) {
    __syncthreads();
    #pragma unroll
    for (int i = 0; i < 16; i++) {
      int e = tid + i * 256;
      int r = e >> 5, nn = e & 31;
      Cs[r * 32 + nn] = (__bf16)((float)Cb[(size_t)r * DSTATE + k0 + nn] * eS[r]);
    }
    #pragma unroll
    for (int i = 0; i < 8; i++) {
      int e = tid + i * 256;
      int p = e >> 5, nn = e & 31;
      Pv[p * 32 + nn] = (__bf16)Pb[(size_t)p * DSTATE + k0 + nn];
    }
    __syncthreads();
    v16bf aC = ld_frag(Cs, wave * 16, lane);
    #pragma unroll
    for (int ni = 0; ni < 4; ni++) {
      v16bf bp = ld_frag(Pv, ni * 16, lane);
      acc[ni] = wmma_bf16(aC, bp, acc[ni]);
    }
  }
  const int rsub = (lane >> 4) * 8, csub = lane & 15;
  float* yb = y + (size_t)bc * CHK * DIN + h * HD;
  #pragma unroll
  for (int ni = 0; ni < 4; ni++)
    #pragma unroll
    for (int r = 0; r < 8; r++) {
      int row = wave * 16 + rsub + r;
      int p = ni * 16 + csub;
      yb[(size_t)row * DIN + p] += acc[ni][r];
    }
}

// ----- gated RMSNorm: ybf = bf16( rmsnorm(y * silu(z)) * w ) ----------------

__global__ __launch_bounds__(256)
void gate_rmsnorm_kernel(const float* __restrict__ zxb, const float* __restrict__ y,
                         const float* __restrict__ nw, __bf16* __restrict__ ybf) {
  const size_t tok = blockIdx.x;
  __shared__ float red[8];
  float vals[4], ss = 0.f;
  #pragma unroll
  for (int i = 0; i < 4; i++) {
    int c = threadIdx.x + i * 256;
    float z = zxb[tok * DIP + c];
    float g = y[tok * DIN + c] * (z / (1.f + expf(-z)));
    vals[i] = g;
    ss += g * g;
  }
  ss = block_sum(ss, red);
  float rms = rsqrtf(ss * (1.f / DIN) + EPSF);
  #pragma unroll
  for (int i = 0; i < 4; i++) {
    int c = threadIdx.x + i * 256;
    ybf[tok * DIN + c] = (__bf16)(vals[i] * rms * nw[c]);
  }
}

// ----- residual add + LayerNorm (512 features) ------------------------------

__global__ __launch_bounds__(256)
void add_ln_kernel(const float* __restrict__ a, const float* __restrict__ r,
                   const float* __restrict__ w, const float* __restrict__ bias,
                   float* __restrict__ outf, __bf16* __restrict__ outbf) {
  const size_t tok = blockIdx.x;
  __shared__ float red[8];
  float v[2], sm = 0.f, sq = 0.f;
  #pragma unroll
  for (int i = 0; i < 2; i++) {
    int c = threadIdx.x + i * 256;
    float x = a[tok * D + c] + r[tok * D + c];
    v[i] = x; sm += x; sq += x * x;
  }
  sm = block_sum(sm, red);
  sq = block_sum(sq, red);
  float mean = sm * (1.f / D);
  float var  = sq * (1.f / D) - mean * mean;
  float inv  = rsqrtf(var + EPSF);
  #pragma unroll
  for (int i = 0; i < 2; i++) {
    int c = threadIdx.x + i * 256;
    float o = (v[i] - mean) * inv * w[c] + bias[c];
    if (outf)  outf[tok * D + c]  = o;
    if (outbf) outbf[tok * D + c] = (__bf16)o;
  }
}

// ---------------------------------------------------------------------------

extern "C" void kernel_launch(void* const* d_in, const int* in_sizes, int n_in,
                              void* d_out, int out_size, void* d_ws, size_t ws_size,
                              hipStream_t stream) {
  (void)in_sizes; (void)n_in; (void)out_size; (void)ws_size;
  const float* tgt        = (const float*)d_in[0];
  const float* in_proj_w  = (const float*)d_in[5];
  const float* conv_w     = (const float*)d_in[6];
  const float* conv_b     = (const float*)d_in[7];
  const float* dt_bias    = (const float*)d_in[8];
  const float* A_log      = (const float*)d_in[9];
  const float* D_head     = (const float*)d_in[10];
  const float* norm_w     = (const float*)d_in[11];
  const float* out_proj_w = (const float*)d_in[12];
  const float* norm1_w    = (const float*)d_in[13];
  const float* norm1_b    = (const float*)d_in[14];
  const float* w1         = (const float*)d_in[15];
  const float* b1         = (const float*)d_in[16];
  const float* w2         = (const float*)d_in[17];
  const float* b2         = (const float*)d_in[18];
  const float* norm3_w    = (const float*)d_in[19];
  const float* norm3_b    = (const float*)d_in[20];
  float* out = (float*)d_out;

  char* ws = (char*)d_ws;
  size_t off = 0;
  auto alloc = [&](size_t bytes) -> void* {
    void* p = ws + off;
    off = (off + bytes + 255) & ~(size_t)255;
    return p;
  };

  __bf16* Xbf   = (__bf16*)alloc((size_t)T * D * 2);
  __bf16* Wip   = (__bf16*)alloc((size_t)DIP * D * 2);
  __bf16* Wop   = (__bf16*)alloc((size_t)D * DIN * 2);
  __bf16* W1b   = (__bf16*)alloc((size_t)FFD * D * 2);
  __bf16* W2b   = (__bf16*)alloc((size_t)D * FFD * 2);
  float*  zxb   = (float*) alloc((size_t)T * DIP * 4);
  __bf16* xbf   = (__bf16*)alloc((size_t)T * DIN * 2);
  __bf16* Bbf   = (__bf16*)alloc((size_t)T * DSTATE * 2);
  __bf16* Cbf   = (__bf16*)alloc((size_t)T * DSTATE * 2);
  float*  dtb   = (float*) alloc((size_t)T * NH * 4);
  float*  acs   = (float*) alloc((size_t)T * NH * 4);
  float*  Gsc   = (float*) alloc((size_t)Bb * NCH * CHK * CHK * 4);
  float*  ybuf  = (float*) alloc((size_t)T * DIN * 4);
  float*  stbuf = (float*) alloc((size_t)Bb * NCH * NH * HD * DSTATE * 4);
  __bf16* ybf   = (__bf16*)alloc((size_t)T * DIN * 2);
  float*  t2    = (float*) alloc((size_t)T * D * 4);
  float*  tbuf  = (float*) alloc((size_t)T * D * 4);
  __bf16* tbf   = (__bf16*)alloc((size_t)T * D * 2);
  __bf16* ffbf  = (__bf16*)alloc((size_t)T * FFD * 2);
  float*  ff2   = (float*) alloc((size_t)T * D * 4);

  auto cvt = [&](const float* src, __bf16* dst, size_t n) {
    cvt_f32_bf16_kernel<<<dim3((unsigned)((n + 255) / 256)), dim3(256), 0, stream>>>(src, dst, n);
  };
  cvt(tgt, Xbf, (size_t)T * D);
  cvt(in_proj_w, Wip, (size_t)DIP * D);
  cvt(out_proj_w, Wop, (size_t)D * DIN);
  cvt(w1, W1b, (size_t)FFD * D);
  cvt(w2, W2b, (size_t)D * FFD);

  // in_proj: (T x 512) @ (2320 x 512)^T
  gemm_nt_kernel<0><<<dim3((DIP + 127) / 128, T / 128), 256, 0, stream>>>(
      Xbf, Wip, nullptr, zxb, T, DIP, D);

  size_t nconv = (size_t)T * CONVD;
  conv_silu_kernel<<<dim3((unsigned)((nconv + 255) / 256)), 256, 0, stream>>>(
      zxb, conv_w, conv_b, xbf, Bbf, Cbf);
  dt_softplus_kernel<<<dim3((T * NH + 255) / 256), 256, 0, stream>>>(zxb, dt_bias, dtb);
  dta_cumsum_kernel<<<dim3((Bb * NCH * NH + 255) / 256), 256, 0, stream>>>(dtb, A_log, acs);

  // SSD
  chunk_scores_kernel<<<dim3(Bb * NCH), 256, 0, stream>>>(Cbf, Bbf, Gsc);
  ydiag_states_kernel<<<dim3(Bb * NCH * NH), 256, 0, stream>>>(
      Gsc, acs, dtb, xbf, Bbf, D_head, ybuf, stbuf);
  scan_kernel<<<dim3(Bb * NH), 256, 0, stream>>>(acs, stbuf);
  yoff_kernel<<<dim3(Bb * NCH * NH), 256, 0, stream>>>(acs, Cbf, stbuf, ybuf);

  // gate + RMSNorm, out_proj
  gate_rmsnorm_kernel<<<dim3(T), 256, 0, stream>>>(zxb, ybuf, norm_w, ybf);
  gemm_nt_kernel<0><<<dim3(D / 128, T / 128), 256, 0, stream>>>(
      ybf, Wop, nullptr, t2, T, D, DIN);

  // residual + LN1 (emit f32 + bf16 copy for FFN)
  add_ln_kernel<<<dim3(T), 256, 0, stream>>>(tgt, t2, norm1_w, norm1_b, tbuf, tbf);

  // FFN: GELU fused into first GEMM epilogue (writes bf16)
  gemm_nt_kernel<1><<<dim3(FFD / 128, T / 128), 256, 0, stream>>>(
      tbf, W1b, b1, ffbf, T, FFD, D);
  gemm_nt_kernel<0><<<dim3(D / 128, T / 128), 256, 0, stream>>>(
      ffbf, W2b, b2, ff2, T, D, FFD);

  // residual + LN3 -> output
  add_ln_kernel<<<dim3(T), 256, 0, stream>>>(tbuf, ff2, norm3_w, norm3_b, out, nullptr);
}
